// MHA_18227841204631
// MI455X (gfx1250) — matrix operand order repaired and checked by reading, not compiled, not run on hardware
//
#include <hip/hip_runtime.h>
#include <hip/hip_bf16.h>
#include <cstdint>
#include <cstddef>

#define DEV __device__ __forceinline__

typedef __attribute__((ext_vector_type(8)))  float  v8f;
typedef __attribute__((ext_vector_type(8)))  __bf16 v8bf;
typedef __attribute__((ext_vector_type(16))) __bf16 v16bf;

typedef __attribute__((ext_vector_type(4))) unsigned int tdm_u32x4;
typedef __attribute__((ext_vector_type(8))) int          tdm_i32x8;
typedef __attribute__((ext_vector_type(4))) int          tdm_i32x4;

static constexpr int D_MODEL  = 768;
static constexpr int N_HEADS  = 12;
static constexpr int HEAD_DIM = 64;
static constexpr int BATCH    = 4;
static constexpr int SEQ      = 2048;

#if __has_builtin(__builtin_amdgcn_tensor_load_to_lds)
#define USE_TDM 1
#else
#define USE_TDM 0
#endif

// ---------------- fragment helpers ----------------

DEV void load8_cvt(const float* p, v16bf& f, int base) {
  v8f v = *(const v8f*)p;                 // 32B load, cvt to bf16 in regs
#pragma unroll
  for (int i = 0; i < 8; ++i) f[base + i] = (__bf16)v[i];
}
DEV void load8_cvt(const __bf16* p, v16bf& f, int base) {
  v8bf v = *(const v8bf*)p;               // 16B load
#pragma unroll
  for (int i = 0; i < 8; ++i) f[base + i] = v[i];
}

DEV v8f wmma_bf16(v16bf a, v16bf b, v8f c) {
  // D = A(16x32) * B(32x16) + C, f32 accum
  return __builtin_amdgcn_wmma_f32_16x16x32_bf16(
      false, a, false, b, (short)0, c, false, false);
}

// A fragment 16x32 (MxK) from row-major A (leading dim lda).
// wave32 layout: lane holds row (lane&15); lanes 0-15 K={0..7,16..23},
// lanes 16-31 K={8..15,24..31}.
template <typename TA>
DEV v16bf frag_a(const TA* A, int lda, int row0, int k0, int lane) {
  const int r    = row0 + (lane & 15);
  const int koff = (lane >> 4) << 3;      // 0 or 8
  const TA* p = A + (size_t)r * lda + k0 + koff;
  v16bf f;
  load8_cvt(p,      f, 0);                // K = k0+koff   .. +7
  load8_cvt(p + 16, f, 8);                // K = k0+koff+16 .. +23
  return f;
}

// B fragment 32x16 (KxN) with B[k][n] = W[n][k]  (W row-major NxK).
// lane holds col (lane&15); lanes 0-15 K={0..7,16..23}, lanes 16-31 K={8..15,24..31}.
template <typename TB>
DEV v16bf frag_b_t(const TB* W, int ldw, int n0, int k0, int lane) {
  const int c    = n0 + (lane & 15);
  const int koff = (lane >> 4) << 3;      // 0 or 8
  const TB* p = W + (size_t)c * ldw + k0 + koff;
  v16bf f;
  load8_cvt(p,     f, 0);
  load8_cvt(p + 16, f, 8);
  return f;
}

DEV void store_c(float* p, float v)  { *p = v; }
DEV void store_c(__bf16* p, float v) { *p = (__bf16)v; }

// ---------------- Tensor Data Mover 2D tile load ----------------
// D# per ISA ch.8: group0 = {flags, lds_addr, global_addr, type=2},
// group1 = dims/strides.  data_size=1 -> 2-byte elements.

#if USE_TDM
DEV void tdm_load_2d_bf16(uint32_t lds_byte_off, const void* gptr,
                          uint32_t tile_w, uint32_t tile_h,
                          uint32_t row_stride_elems) {
  const uint64_t ga = (uint64_t)(uintptr_t)gptr;
  tdm_u32x4 g0;
  g0[0] = 1u;                                            // count=1 (valid)
  g0[1] = lds_byte_off;                                  // lds_addr
  g0[2] = (uint32_t)ga;                                  // global_addr[31:0]
  g0[3] = (uint32_t)((ga >> 32) & 0x01FFFFFFu)           // global_addr[56:32]
          | (2u << 30);                                  // type=2 ("image")
  tdm_i32x8 g1;
  g1[0] = (int)(1u << 16);            // wg_mask=0, data_size=1 (2B)
  g1[1] = (int)(tile_w << 16);        // tensor_dim0[15:0]  (bits 63:48)
  g1[2] = (int)(tile_h << 16);        // tensor_dim1[15:0]  (bits 95:80)
  g1[3] = (int)(tile_w << 16);        // tile_dim0          (bits 127:112)
  g1[4] = (int)tile_h;                // tile_dim1 (tile_dim2=0)
  g1[5] = (int)row_stride_elems;      // tensor_dim0_stride[31:0]
  g1[6] = 0;                          // stride hi, dim1_stride lo
  g1[7] = 0;
  tdm_i32x4 z = {};
#if __clang_major__ >= 23
  tdm_i32x8 z8 = {};
  __builtin_amdgcn_tensor_load_to_lds(g0, g1, z, z, z8, 0);
#else
  __builtin_amdgcn_tensor_load_to_lds(g0, g1, z, z, 0);
#endif
}
#endif

// ---------------- GEMM: C[M,N] = A[M,K] @ W[N,K]^T + bias ----------------
// 8 waves/block, wave tile 32x32 (2x2 WMMA), block tile 128x64.

template <typename TA, typename TC>
__global__ __launch_bounds__(256) void gemm_nt_bias(
    const TA* __restrict__ A, const float* __restrict__ W,
    const float* __restrict__ bias, TC* __restrict__ C,
    int M, int N, int K) {
  const int lane = threadIdx.x & 31;
  const int wave = threadIdx.x >> 5;
  const int m0 = blockIdx.x * 128 + (wave & 3) * 32;
  const int n0 = blockIdx.y * 64  + (wave >> 2) * 32;

  v8f acc[2][2] = {};
  for (int k = 0; k < K; k += 32) {
    if (k + 32 < K) {   // warm GL2 for the next K-slab (global_prefetch_b8)
      __builtin_prefetch(A + (size_t)(m0 + (lane & 15)) * K + k + 32, 0, 0);
      __builtin_prefetch(W + (size_t)(n0 + (lane & 15)) * K + k + 32, 0, 0);
    }
    v16bf a0 = frag_a(A, K, m0,      k, lane);
    v16bf a1 = frag_a(A, K, m0 + 16, k, lane);
    v16bf b0 = frag_b_t(W, K, n0,      k, lane);
    v16bf b1 = frag_b_t(W, K, n0 + 16, k, lane);
    acc[0][0] = wmma_bf16(a0, b0, acc[0][0]);
    acc[0][1] = wmma_bf16(a0, b1, acc[0][1]);
    acc[1][0] = wmma_bf16(a1, b0, acc[1][0]);
    acc[1][1] = wmma_bf16(a1, b1, acc[1][1]);
  }

  // C layout: lane holds col (lane&15); VGPR e is row e (+8 for lanes>=16)
  const int nl = lane & 15;
  const int mo = (lane >> 4) << 3;
#pragma unroll
  for (int mi = 0; mi < 2; ++mi)
#pragma unroll
    for (int ni = 0; ni < 2; ++ni) {
      const int   n  = n0 + ni * 16 + nl;
      const float bv = bias[n];
      const int   mb = m0 + mi * 16 + mo;
#pragma unroll
      for (int e = 0; e < 8; ++e)
        store_c(C + (size_t)(mb + e) * N + n, acc[mi][ni][e] + bv);
    }
}

// ---------------- Flash attention ----------------
// grid = (SEQ/128, N_HEADS, BATCH); 8 waves x 16 queries each.
// Computes S^T = K_tile @ q^T so exp(S) lands directly in the A-fragment
// layout for the P @ V WMMA (no cross-lane transpose).
// K/V tiles double-buffered: the Tensor Data Mover streams tile i+1 while
// the SIMDs run WMMAs on tile i (TENSORcnt software pipeline).

__global__ __launch_bounds__(256) void flash_attn(
    const __bf16* __restrict__ Qp, const __bf16* __restrict__ Kp,
    const __bf16* __restrict__ Vp, __bf16* __restrict__ Op) {
  constexpr int KB = 32;
#if USE_TDM
  constexpr int NB = 2;                     // double buffer
#else
  constexpr int NB = 1;
#endif
  __shared__ __bf16 s_k [NB][KB][HEAD_DIM]; // keys row-major (TDM target)
  __shared__ __bf16 s_v [NB][KB][HEAD_DIM]; // V row-major (TDM target)
  __shared__ __bf16 s_vt[NB][HEAD_DIM][KB]; // V transposed: [d][key]

  const int lane = threadIdx.x & 31;
  const int wave = threadIdx.x >> 5;
  const int b = blockIdx.z, h = blockIdx.y;
  const int m0 = blockIdx.x * 128 + wave * 16;

  const size_t base = (size_t)b * SEQ * D_MODEL + (size_t)h * HEAD_DIM;
  const __bf16* qb  = Qp + base;
  const __bf16* kbp = Kp + base;
  const __bf16* vbp = Vp + base;

  // q as the B operand of S^T: lane = query column, d-slices {0..7,16..23}/{8..15,24..31}.
  v16bf qf[2];
  {
    const int qi   = lane & 15;
    const int koff = (lane >> 4) << 3;
    const __bf16* p = qb + (size_t)(m0 + qi) * D_MODEL;
#pragma unroll
    for (int t = 0; t < 2; ++t) {
      const __bf16* pp = p + t * 32 + koff;
      load8_cvt(pp,     qf[t], 0);
      load8_cvt(pp + 16, qf[t], 8);
    }
  }

  v8f oacc[4] = {};
  float m_run = -3.402823466e+38f;
  float l_run = 0.f;
  const float scale = 0.125f;             // 1/sqrt(64)

  const int tr = threadIdx.x >> 3;        // staging row 0..31
  const int tc = (threadIdx.x & 7) * 8;   // staging col 0..56

#if USE_TDM
  // prologue: kick off DMA for tile 0 (no wait -> overlaps q-frag loads)
  if (wave == 0) {
    tdm_load_2d_bf16((uint32_t)(uintptr_t)&s_k[0][0][0], kbp,
                     HEAD_DIM, KB, D_MODEL);
    tdm_load_2d_bf16((uint32_t)(uintptr_t)&s_v[0][0][0], vbp,
                     HEAD_DIM, KB, D_MODEL);
  }
#endif

  int cur = 0;
  for (int k0 = 0; k0 < SEQ; k0 += KB) {
#if USE_TDM
    if (wave == 0) __builtin_amdgcn_s_wait_tensorcnt(0);  // tile k0 landed
    __syncthreads();                                      // publish tile k0
    if (wave == 0 && k0 + KB < SEQ) {                     // overlap: tile k0+KB
      const int nxt = cur ^ 1;
      tdm_load_2d_bf16((uint32_t)(uintptr_t)&s_k[nxt][0][0],
                       kbp + (size_t)(k0 + KB) * D_MODEL, HEAD_DIM, KB, D_MODEL);
      tdm_load_2d_bf16((uint32_t)(uintptr_t)&s_v[nxt][0][0],
                       vbp + (size_t)(k0 + KB) * D_MODEL, HEAD_DIM, KB, D_MODEL);
    }
    {   // transpose V tile: s_v[cur] -> s_vt[cur]
      v8bf vv = *(const v8bf*)&s_v[cur][tr][tc];
#pragma unroll
      for (int i = 0; i < 8; ++i) s_vt[cur][tc + i][tr] = vv[i];
    }
#else
    __syncthreads();
    {
      v8bf kv = *(const v8bf*)(kbp + (size_t)(k0 + tr) * D_MODEL + tc);
      *(v8bf*)&s_k[cur][tr][tc] = kv;
      v8bf vv = *(const v8bf*)(vbp + (size_t)(k0 + tr) * D_MODEL + tc);
#pragma unroll
      for (int i = 0; i < 8; ++i) s_vt[cur][tc + i][tr] = vv[i];
    }
#endif
    __syncthreads();

    // S^T: keys 0-15 -> s0, keys 16-31 -> s1 (each K=64 via 2 WMMAs)
    v8f s0 = {}, s1 = {};
    {
      const int kr   = lane & 15;
      const int koff = (lane >> 4) << 3;
#pragma unroll
      for (int t = 0; t < 2; ++t) {
        v16bf ka, kc;
        load8_cvt(&s_k[cur][kr     ][t * 32 + koff     ], ka, 0);
        load8_cvt(&s_k[cur][kr     ][t * 32 + koff + 16], ka, 8);
        load8_cvt(&s_k[cur][16 + kr][t * 32 + koff     ], kc, 0);
        load8_cvt(&s_k[cur][16 + kr][t * 32 + koff + 16], kc, 8);
        s0 = wmma_bf16(ka, qf[t], s0);
        s1 = wmma_bf16(kc, qf[t], s1);
      }
    }

    // online softmax; per-lane query = lane&15 (groups symmetric via xor16)
    float mb = -3.402823466e+38f;
#pragma unroll
    for (int e = 0; e < 8; ++e) {
      s0[e] *= scale; s1[e] *= scale;
      mb = fmaxf(mb, fmaxf(s0[e], s1[e]));
    }
    mb = fmaxf(mb, __shfl_xor(mb, 16, 32));
    const float m_new = fmaxf(m_run, mb);
    const float alpha = __expf(m_run - m_new);
    float p0[8], p1[8], lb = 0.f;
#pragma unroll
    for (int e = 0; e < 8; ++e) {
      p0[e] = __expf(s0[e] - m_new);
      p1[e] = __expf(s1[e] - m_new);
      lb += p0[e] + p1[e];
    }
    lb += __shfl_xor(lb, 16, 32);
    l_run = l_run * alpha + lb;
    m_run = m_new;

    // P packs straight into a 16x32 A-fragment (layout identity)
    v16bf pa;
#pragma unroll
    for (int e = 0; e < 8; ++e) { pa[e] = (__bf16)p0[e]; pa[8 + e] = (__bf16)p1[e]; }

    // rescale O accumulators by per-query alpha (query = e + 8*(lane>=16))
    const int qsel = (lane & 16) >> 1;
#pragma unroll
    for (int e = 0; e < 8; ++e) {
      const float al = __shfl(alpha, e + qsel, 32);
#pragma unroll
      for (int t = 0; t < 4; ++t) oacc[t][e] *= al;
    }

    // O += P(16x32) @ V(32x64): 4 d-tiles of 16, B frags from s_vt
#pragma unroll
    for (int t = 0; t < 4; ++t) {
      const int d     = t * 16 + (lane & 15);
      const int koff2 = (lane >> 4) << 3;
      v16bf vf;
      load8_cvt(&s_vt[cur][d][koff2    ], vf, 0);
      load8_cvt(&s_vt[cur][d][koff2 + 16], vf, 8);
      oacc[t] = wmma_bf16(pa, vf, oacc[t]);
    }
    cur ^= (NB - 1);
  }

  // normalize by 1/l and store [B,S,D] bf16
  const int qsel = (lane & 16) >> 1;
  __bf16* ob = Op + base;
#pragma unroll
  for (int e = 0; e < 8; ++e) {
    const float il  = 1.0f / __shfl(l_run, e + qsel, 32);
    const size_t row = (size_t)(m0 + e + qsel) * D_MODEL;
#pragma unroll
    for (int t = 0; t < 4; ++t)
      ob[row + t * 16 + (lane & 15)] = (__bf16)(oacc[t][e] * il);
  }
}

// ---------------- launch ----------------

extern "C" void kernel_launch(void* const* d_in, const int* in_sizes, int n_in,
                              void* d_out, int out_size, void* d_ws, size_t ws_size,
                              hipStream_t stream) {
  (void)in_sizes; (void)n_in; (void)out_size; (void)ws_size;
  const float* Q  = (const float*)d_in[0];
  const float* K  = (const float*)d_in[1];
  const float* V  = (const float*)d_in[2];
  const float* Wq = (const float*)d_in[3];
  const float* bq = (const float*)d_in[4];
  const float* Wk = (const float*)d_in[5];
  const float* bk = (const float*)d_in[6];
  const float* Wv = (const float*)d_in[7];
  const float* bv = (const float*)d_in[8];
  const float* Wo = (const float*)d_in[9];
  const float* bo = (const float*)d_in[10];

  const int M = BATCH * SEQ;                       // 8192
  const size_t act = (size_t)M * D_MODEL;          // elems per activation
  char* ws = (char*)d_ws;
  __bf16* qp = (__bf16*)(ws + 0 * act * 2);
  __bf16* kp = (__bf16*)(ws + 1 * act * 2);
  __bf16* vp = (__bf16*)(ws + 2 * act * 2);
  __bf16* ap = (__bf16*)(ws + 3 * act * 2);        // total 50.3 MB

  dim3 blk(256);
  dim3 gg(M / 128, D_MODEL / 64);                  // 64 x 12

  gemm_nt_bias<float, __bf16><<<gg, blk, 0, stream>>>(Q, Wq, bq, qp, M, D_MODEL, D_MODEL);
  gemm_nt_bias<float, __bf16><<<gg, blk, 0, stream>>>(K, Wk, bk, kp, M, D_MODEL, D_MODEL);
  gemm_nt_bias<float, __bf16><<<gg, blk, 0, stream>>>(V, Wv, bv, vp, M, D_MODEL, D_MODEL);

  flash_attn<<<dim3(SEQ / 128, N_HEADS, BATCH), blk, 0, stream>>>(qp, kp, vp, ap);

  gemm_nt_bias<__bf16, float><<<gg, blk, 0, stream>>>(ap, Wo, bo, (float*)d_out, M, D_MODEL, D_MODEL);
}
